// SphereNet_39015482917557
// MI455X (gfx1250) — matrix operand order repaired and checked by reading, not compiled, not run on hardware
//
#include <hip/hip_runtime.h>
#include <math.h>

typedef _Float16 half_t;
typedef __attribute__((ext_vector_type(16))) _Float16 v16h;
typedef __attribute__((ext_vector_type(8)))  _Float16 v8h;
typedef __attribute__((ext_vector_type(8)))  float    v8f;
typedef __attribute__((ext_vector_type(4)))  float    v4f;

#define PI_F 3.14159265358979323846f

__device__ __forceinline__ float silu_f(float x) {
    return x / (1.0f + __expf(-x));
}

// ---------------------------------------------------------------------------
// Utility kernels
// ---------------------------------------------------------------------------
__global__ void zero_kernel(float* p, size_t n) {
    size_t i = (size_t)blockIdx.x * blockDim.x + threadIdx.x;
    if (i < n) p[i] = 0.0f;
}

// Pack a row-major fp32 weight W[K,N] into WMMA B-fragment order (f16):
//   Wp[(((tn*(K/32) + ks)*32 + lane)*16 + h] = W[k, col]
//   col = tn*16 + (lane&15),  k = ks*32 + ((lane>>4)<<3) + (h&7) + ((h>>3)<<4)
// so each lane's 16 halves per k-step are one contiguous 32-byte chunk.
__global__ void pack_w_kernel(const float* __restrict__ W, half_t* __restrict__ Wp,
                              int K, int N) {
    int p = blockIdx.x * blockDim.x + threadIdx.x;
    if (p >= K * N) return;
    int h = p & 15;
    int lane = (p >> 4) & 31;
    int rest = p >> 9;                 // / (32*16)
    int kst = K >> 5;
    int ks = rest % kst;
    int tn = rest / kst;
    int col = tn * 16 + (lane & 15);
    int k = ks * 32 + ((lane >> 4) << 3) + (h & 7) + ((h >> 3) << 4);
    Wp[p] = (half_t)W[(size_t)k * N + col];
}

// ---------------------------------------------------------------------------
// Geometry: per-edge distance + Bessel radial basis with polynomial envelope
// ---------------------------------------------------------------------------
__global__ void edge_geom_kernel(const float* __restrict__ pos,
                                 const int* __restrict__ ei,
                                 const int* __restrict__ ej,
                                 float* __restrict__ dist,
                                 float* __restrict__ rbf, int E) {
    int e = blockIdx.x * blockDim.x + threadIdx.x;
    if (e >= E) return;
    int i = ei[e], j = ej[e];
    float dx = pos[i * 3 + 0] - pos[j * 3 + 0];
    float dy = pos[i * 3 + 1] - pos[j * 3 + 1];
    float dz = pos[i * 3 + 2] - pos[j * 3 + 2];
    float d = sqrtf(dx * dx + dy * dy + dz * dz);
    dist[e] = d;
    float x = d * (1.0f / 5.0f);             // d / CUTOFF
    float x4 = x * x; x4 = x4 * x4;          // x^(p-1), p=5
    float env = 1.0f / x + (-21.0f) * x4 + 35.0f * x4 * x + (-15.0f) * x4 * x * x;
#pragma unroll
    for (int r = 0; r < 6; ++r)
        rbf[(size_t)e * 6 + r] = env * sinf((float)(r + 1) * PI_F * x);
}

__global__ void angle_kernel(const float* __restrict__ pos,
                             const int* __restrict__ ti,
                             const int* __restrict__ tj,
                             const int* __restrict__ tk,
                             float* __restrict__ angle,
                             unsigned int* __restrict__ torsu, int T) {
    int t = blockIdx.x * blockDim.x + threadIdx.x;
    if (t >= T) return;
    int i = ti[t], j = tj[t], k = tk[t];
    float jx = pos[j * 3 + 0], jy = pos[j * 3 + 1], jz = pos[j * 3 + 2];
    float ax = pos[i * 3 + 0] - jx, ay = pos[i * 3 + 1] - jy, az = pos[i * 3 + 2] - jz;
    float bx = pos[k * 3 + 0] - jx, by = pos[k * 3 + 1] - jy, bz = pos[k * 3 + 2] - jz;
    float cx = ay * bz - az * by;
    float cy = az * bx - ax * bz;
    float cz = ax * by - ay * bx;
    float cn = sqrtf(cx * cx + cy * cy + cz * cz);
    float dp = ax * bx + ay * by + az * bz;
    angle[t] = atan2f(cn, dp);
    torsu[t] = 0x7f800000u;  // +inf seed for segment-min
}

// per-quadruplet dihedral; segment-min via bitwise atomicMin (values in (0,2pi])
__global__ void torsion_kernel(const float* __restrict__ pos,
                               const int* __restrict__ qi, const int* __restrict__ qj,
                               const int* __restrict__ qk, const int* __restrict__ qkn,
                               const int* __restrict__ qtri,
                               unsigned int* __restrict__ torsu, int Q) {
    int q = blockIdx.x * blockDim.x + threadIdx.x;
    if (q >= Q) return;
    int i = qi[q], j = qj[q], k = qk[q], kn = qkn[q];
    float jx = pos[j * 3 + 0], jy = pos[j * 3 + 1], jz = pos[j * 3 + 2];
    float p0x = pos[k * 3 + 0] - jx, p0y = pos[k * 3 + 1] - jy, p0z = pos[k * 3 + 2] - jz;
    float pix = pos[i * 3 + 0] - jx, piy = pos[i * 3 + 1] - jy, piz = pos[i * 3 + 2] - jz;
    float pkx = pos[kn * 3 + 0] - jx, pky = pos[kn * 3 + 1] - jy, pkz = pos[kn * 3 + 2] - jz;
    float dji = sqrtf(pix * pix + piy * piy + piz * piz);
    float a1x = piy * p0z - piz * p0y;
    float a1y = piz * p0x - pix * p0z;
    float a1z = pix * p0y - piy * p0x;
    float a2x = piy * pkz - piz * pky;
    float a2y = piz * pkx - pix * pkz;
    float a2z = pix * pky - piy * pkx;
    float a = a1x * a2x + a1y * a2y + a1z * a2z;
    float crx = a1y * a2z - a1z * a2y;
    float cry = a1z * a2x - a1x * a2z;
    float crz = a1x * a2y - a1y * a2x;
    float b = (crx * pix + cry * piy + crz * piz) / dji;
    float tors = atan2f(b, a);
    if (tors <= 0.0f) tors += 2.0f * PI_F;
    atomicMin(&torsu[qtri[q]], __float_as_uint(tors));
}

// ---------------------------------------------------------------------------
// Small-K fused scalar kernels (K = 6 or 8 — too small for WMMA's K=32)
// ---------------------------------------------------------------------------
// rbf0h = silu(rbf @ rbf0_W + rbf0_b)  -> f16 [E,128]
__global__ void rbf0h_kernel(const float* __restrict__ rbf,
                             const float* __restrict__ W, const float* __restrict__ b,
                             half_t* __restrict__ out, int E) {
    size_t idx = (size_t)blockIdx.x * blockDim.x + threadIdx.x;
    if (idx >= (size_t)E * 128) return;
    int e = (int)(idx >> 7), c = (int)(idx & 127);
    const float* rr = rbf + (size_t)e * 6;
    float s = b[c];
#pragma unroll
    for (int r = 0; r < 6; ++r) s += rr[r] * W[r * 128 + c];
    out[idx] = (half_t)silu_f(s);
}

// out_f32 = (rbf @ W6x128) * e1_f16   (init_rbf1 / e_rbf gates; feeds f32 atomics)
__global__ void mulrbf_kernel(const float* __restrict__ rbf,
                              const float* __restrict__ W,
                              const half_t* __restrict__ e1,
                              float* __restrict__ out, int E) {
    size_t idx = (size_t)blockIdx.x * blockDim.x + threadIdx.x;
    if (idx >= (size_t)E * 128) return;
    int e = (int)(idx >> 7), c = (int)(idx & 127);
    const float* rr = rbf + (size_t)e * 6;
    float s = 0.0f;
#pragma unroll
    for (int r = 0; r < 6; ++r) s += rr[r] * W[r * 128 + c];
    out[idx] = s * (float)e1[idx];
}

// x_kj *= (rbf @ e_rbf1[6x8]) @ e_rbf2[8x128]   (in place, f16)
__global__ void mulrbfh_kernel(const float* __restrict__ rbf,
                               const float* __restrict__ W1,
                               const float* __restrict__ W2,
                               half_t* __restrict__ xkj, int E) {
    size_t idx = (size_t)blockIdx.x * blockDim.x + threadIdx.x;
    if (idx >= (size_t)E * 128) return;
    int e = (int)(idx >> 7), c = (int)(idx & 127);
    const float* rr = rbf + (size_t)e * 6;
    float s = 0.0f;
#pragma unroll
    for (int j = 0; j < 8; ++j) {
        float t = 0.0f;
#pragma unroll
        for (int r = 0; r < 6; ++r) t += rr[r] * W1[r * 8 + j];
        s += t * W2[j * 128 + c];
    }
    xkj[idx] = (half_t)((float)xkj[idx] * s);
}

// segment_sum of e2[E,128] into nodeacc[N,128] by edge_i
__global__ void scatter_node_kernel(const float* __restrict__ e2,
                                    const int* __restrict__ ei,
                                    float* __restrict__ nodeacc, int E) {
    size_t idx = (size_t)blockIdx.x * blockDim.x + threadIdx.x;
    if (idx >= (size_t)E * 128) return;
    int e = (int)(idx >> 7), c = (int)(idx & 127);
    atomicAdd(&nodeacc[(size_t)ei[e] * 128 + c], e2[idx]);
}

__global__ void vout_kernel(const float* __restrict__ v,
                            const float* __restrict__ w,
                            float* __restrict__ vtot, int N) {
    int n = blockIdx.x * blockDim.x + threadIdx.x;
    if (n >= N) return;
    const float* vr = v + (size_t)n * 256;
    float s = 0.0f;
    for (int c = 0; c < 256; ++c) s += vr[c] * w[c];
    vtot[n] += s;
}

__global__ void batch_kernel(const float* __restrict__ vtot,
                             const int* __restrict__ batch,
                             float* __restrict__ out, int N) {
    int n = blockIdx.x * blockDim.x + threadIdx.x;
    if (n >= N) return;
    atomicAdd(&out[batch[n]], vtot[n]);
}

// ---------------------------------------------------------------------------
// Triplet message kernel: fused sbf/tbf basis -> bottleneck -> gated message
// -> scatter-add into agg[idx_ji]
// ---------------------------------------------------------------------------
__global__ void triplet_kernel(const float* __restrict__ angle,
                               const unsigned int* __restrict__ torsu,
                               const float* __restrict__ rbf,
                               const float* __restrict__ xkj2,
                               const int* __restrict__ idx_kj,
                               const int* __restrict__ idx_ji,
                               const float* __restrict__ e_sbf1,  // [18,8]
                               const float* __restrict__ e_sbf2,  // [8,64]
                               const float* __restrict__ e_t1,    // [54,8]
                               const float* __restrict__ e_t2,    // [8,64]
                               float* __restrict__ agg, int T) {
    int t = blockIdx.x * blockDim.x + threadIdx.x;
    if (t >= T) return;
    int ekj = idx_kj[t], eji = idx_ji[t];
    float rb[6];
#pragma unroll
    for (int r = 0; r < 6; ++r) rb[r] = rbf[(size_t)ekj * 6 + r];
    float a = angle[t];
    float to = __uint_as_float(torsu[t]);
    float ca[3] = {1.0f, cosf(a), cosf(2.0f * a)};
    float ct[3] = {1.0f, cosf(to), cosf(2.0f * to)};
    float cc[9];
#pragma unroll
    for (int l = 0; l < 3; ++l)
#pragma unroll
        for (int m = 0; m < 3; ++m) cc[l * 3 + m] = ca[l] * ct[m];
    float s8[8], t8[8];
#pragma unroll
    for (int j = 0; j < 8; ++j) {
        float s = 0.0f;
#pragma unroll
        for (int l = 0; l < 3; ++l)
#pragma unroll
            for (int r = 0; r < 6; ++r)
                s += ca[l] * rb[r] * e_sbf1[(l * 6 + r) * 8 + j];
        s8[j] = s;
        float tv = 0.0f;
        for (int lm = 0; lm < 9; ++lm)
#pragma unroll
            for (int r = 0; r < 6; ++r)
                tv += cc[lm] * rb[r] * e_t1[(lm * 6 + r) * 8 + j];
        t8[j] = tv;
    }
    const float* xr = xkj2 + (size_t)ekj * 64;
    float* ar = agg + (size_t)eji * 64;
    for (int c = 0; c < 64; ++c) {
        float sv = 0.0f, tv = 0.0f;
#pragma unroll
        for (int j = 0; j < 8; ++j) {
            sv += s8[j] * e_sbf2[j * 64 + c];
            tv += t8[j] * e_t2[j * 64 + c];
        }
        atomicAdd(&ar[c], xr[c] * sv * tv);
    }
}

// ---------------------------------------------------------------------------
// WMMA GEMM with fragment-packed f16 weights.
// One wave owns a 16-row strip: A-fragments (K/32 of them) loaded ONCE into
// registers, then reused across all N-tiles. B fragments are single contiguous
// 32B vector loads from the packed weight layout.
// Fragment layouts per CDNA5 ISA 7.12.2 (wave32):
//   A: lane L -> M = L&15; halves h<8 -> K=kb+h, h>=8 -> K=kb+16+(h-8), kb=(L>>4)*8
//   B: lane L -> N = L&15; half h -> K as for A (same kb split)
//   C/D: VGPR r, lane L -> M = (L>>4)*8 + r, N = L&15
// rows must be a multiple of 16 (true for all call sites).
// ---------------------------------------------------------------------------
__device__ __forceinline__ v16h load_a_frag(const half_t* Ar, int kb) {
    v8h lo = *(const v8h*)(Ar + kb);
    v8h hi = *(const v8h*)(Ar + kb + 16);
    v16h av;
#pragma unroll
    for (int h = 0; h < 8; ++h) { av[h] = lo[h]; av[h + 8] = hi[h]; }
    return av;
}
__device__ __forceinline__ v16h load_a_frag(const float* Ar, int kb) {
    v4f f0 = *(const v4f*)(Ar + kb);
    v4f f1 = *(const v4f*)(Ar + kb + 4);
    v4f f2 = *(const v4f*)(Ar + kb + 16);
    v4f f3 = *(const v4f*)(Ar + kb + 20);
    v16h av;
#pragma unroll
    for (int h = 0; h < 4; ++h) {
        av[h] = (half_t)f0[h];
        av[h + 4] = (half_t)f1[h];
        av[h + 8] = (half_t)f2[h];
        av[h + 12] = (half_t)f3[h];
    }
    return av;
}
__device__ __forceinline__ void store_c(float* C, size_t i, float v) { C[i] = v; }
__device__ __forceinline__ void store_c(half_t* C, size_t i, float v) { C[i] = (half_t)v; }
__device__ __forceinline__ float load_add(const float* A, size_t i) { return A[i]; }
__device__ __forceinline__ float load_add(const half_t* A, size_t i) { return (float)A[i]; }

template <int K, bool SILU, bool HAS_BIAS, bool HAS_ADD, typename AT, typename OT>
__global__ __launch_bounds__(256) void gemm_pk_kernel(
    const AT* __restrict__ A, const half_t* __restrict__ Wp,
    const float* __restrict__ bias, const OT* __restrict__ add,
    OT* __restrict__ C, int mtiles, int Ncols) {
    constexpr int KS = K / 32;
    const int lane = threadIdx.x & 31;
    const int wave = threadIdx.x >> 5;
    const int ntiles = Ncols >> 4;
    const int kA0 = (lane >> 4) << 3;
    for (int tm = blockIdx.x * 8 + wave; tm < mtiles; tm += gridDim.x * 8) {
        const AT* Ar = A + (size_t)(tm * 16 + (lane & 15)) * K;
        v16h afrag[KS];
#pragma unroll
        for (int ks = 0; ks < KS; ++ks) afrag[ks] = load_a_frag(Ar, ks * 32 + kA0);
        for (int tn = 0; tn < ntiles; ++tn) {
            const half_t* Bp = Wp + ((size_t)(tn * KS) * 32 + lane) * 16;
            v8f acc = {};
#pragma unroll
            for (int ks = 0; ks < KS; ++ks) {
                v16h bv = *(const v16h*)(Bp + (size_t)ks * 32 * 16);
                acc = __builtin_amdgcn_wmma_f32_16x16x32_f16(
                    false, afrag[ks], false, bv, (short)0, acc, false, false);
            }
            const int col = tn * 16 + (lane & 15);
#pragma unroll
            for (int r = 0; r < 8; ++r) {
                int m = tm * 16 + ((lane >> 4) << 3) + r;
                size_t ci = (size_t)m * Ncols + col;
                float v = acc[r];
                if (HAS_BIAS) v += bias[col];
                if (SILU) v = silu_f(v);
                if (HAS_ADD) v += load_add(add, ci);
                store_c(C, ci, v);
            }
        }
    }
}

// Init-block GEMM: e1 = silu(concat[h[edge_i], h[edge_j], rbf0h] @ W[384,128] + b)
// 384-wide A is virtual: gathered from fp32 emb_table + f16 rbf0h at load time.
__global__ __launch_bounds__(256) void gemm_init_kernel(
    const float* __restrict__ emb, const int* __restrict__ xatom,
    const int* __restrict__ ei, const int* __restrict__ ej,
    const half_t* __restrict__ rbf0h, const half_t* __restrict__ Wp,
    const float* __restrict__ bias, half_t* __restrict__ C, int mtiles) {
    constexpr int K = 384, KS = 12, NT = 8, NC = 128;
    const int lane = threadIdx.x & 31;
    const int wave = threadIdx.x >> 5;
    const int kA0 = (lane >> 4) << 3;
    for (int tm = blockIdx.x * 8 + wave; tm < mtiles; tm += gridDim.x * 8) {
        const int arow = tm * 16 + (lane & 15);
        const float* ha = emb + (size_t)xatom[ei[arow]] * 128;
        const float* hb = emb + (size_t)xatom[ej[arow]] * 128;
        const half_t* hr = rbf0h + (size_t)arow * 128;
        v16h afrag[KS];
#pragma unroll
        for (int ks = 0; ks < KS; ++ks) {
#pragma unroll
            for (int h = 0; h < 8; ++h) {
                int k0 = ks * 32 + kA0 + h;
                int k1 = k0 + 16;
                afrag[ks][h] = (k0 < 128) ? (half_t)ha[k0]
                             : (k0 < 256) ? (half_t)hb[k0 - 128] : hr[k0 - 256];
                afrag[ks][h + 8] = (k1 < 128) ? (half_t)ha[k1]
                               : (k1 < 256) ? (half_t)hb[k1 - 128] : hr[k1 - 256];
            }
        }
        for (int tn = 0; tn < NT; ++tn) {
            const half_t* Bp = Wp + ((size_t)(tn * KS) * 32 + lane) * 16;
            v8f acc = {};
#pragma unroll
            for (int ks = 0; ks < KS; ++ks) {
                v16h bv = *(const v16h*)(Bp + (size_t)ks * 32 * 16);
                acc = __builtin_amdgcn_wmma_f32_16x16x32_f16(
                    false, afrag[ks], false, bv, (short)0, acc, false, false);
            }
            const int col = tn * 16 + (lane & 15);
#pragma unroll
            for (int r = 0; r < 8; ++r) {
                int m = tm * 16 + ((lane >> 4) << 3) + r;
                C[(size_t)m * NC + col] = (half_t)silu_f(acc[r] + bias[col]);
            }
        }
    }
}

// ---------------------------------------------------------------------------
// Host-side orchestration
// ---------------------------------------------------------------------------
extern "C" void kernel_launch(void* const* d_in, const int* in_sizes, int n_in,
                              void* d_out, int out_size, void* d_ws, size_t ws_size,
                              hipStream_t stream) {
    const float* pos          = (const float*)d_in[0];
    const float* p_emb        = (const float*)d_in[1];   // [95,128]
    const float* p_rbf0_W     = (const float*)d_in[2];   // [6,128]
    const float* p_rbf0_b     = (const float*)d_in[3];   // [128]
    const float* p_initlin_W  = (const float*)d_in[4];   // [384,128]
    const float* p_initlin_b  = (const float*)d_in[5];   // [128]
    const float* p_initrbf1_W = (const float*)d_in[6];   // [6,128]
    const float* p_e_rbf1     = (const float*)d_in[7];   // [6,8]
    const float* p_e_rbf2     = (const float*)d_in[8];   // [8,128]
    const float* p_e_sbf1     = (const float*)d_in[9];   // [18,8]
    const float* p_e_sbf2     = (const float*)d_in[10];  // [8,64]
    const float* p_e_t1       = (const float*)d_in[11];  // [54,8]
    const float* p_e_t2       = (const float*)d_in[12];  // [8,64]
    const float* p_e_rbf      = (const float*)d_in[13];  // [6,128]
    const float* p_e_kj_W     = (const float*)d_in[14];
    const float* p_e_kj_b     = (const float*)d_in[15];
    const float* p_e_ji_W     = (const float*)d_in[16];
    const float* p_e_ji_b     = (const float*)d_in[17];
    const float* p_e_down     = (const float*)d_in[18];  // [128,64]
    const float* p_e_up       = (const float*)d_in[19];  // [64,128]
    const float* p_resb_W     = (const float*)d_in[20];  // [1,2,128,128]
    const float* p_resb_b     = (const float*)d_in[21];  // [1,2,128]
    const float* p_e_lin_W    = (const float*)d_in[22];
    const float* p_e_lin_b    = (const float*)d_in[23];
    const float* p_resa_W     = (const float*)d_in[24];  // [2,2,128,128]
    const float* p_resa_b     = (const float*)d_in[25];  // [2,2,128]
    const float* p_v_up_W     = (const float*)d_in[26];  // [128,256]
    const float* p_v_up_b     = (const float*)d_in[27];  // [256]
    const float* p_v_lins_W   = (const float*)d_in[28];  // [3,256,256]
    const float* p_v_lins_b   = (const float*)d_in[29];  // [3,256]
    const float* p_v_out_W    = (const float*)d_in[30];  // [256,1]
    const int* x_atom = (const int*)d_in[31];
    const int* edge_i = (const int*)d_in[32];
    const int* edge_j = (const int*)d_in[33];
    const int* t_i    = (const int*)d_in[34];
    const int* t_j    = (const int*)d_in[35];
    const int* t_k    = (const int*)d_in[36];
    const int* idx_kj = (const int*)d_in[37];
    const int* idx_ji = (const int*)d_in[38];
    const int* q_i    = (const int*)d_in[39];
    const int* q_j    = (const int*)d_in[40];
    const int* q_k    = (const int*)d_in[41];
    const int* q_kn   = (const int*)d_in[42];
    const int* q_tri  = (const int*)d_in[43];
    const int* batch  = (const int*)d_in[44];

    const int N = in_sizes[31];
    const int E = in_sizes[32];
    const int T = in_sizes[34];
    const int Q = in_sizes[39];
    float* out = (float*)d_out;

    // ---- workspace carving ----
    char* ws = (char*)d_ws;
    size_t off = 0;
    auto alloc = [&](size_t bytes) -> void* {
        void* p = ws + off;
        off += (bytes + 255) & ~(size_t)255;
        return p;
    };
    // fragment-packed f16 weights
    half_t* w_initlin = (half_t*)alloc((size_t)384 * 128 * sizeof(half_t));
    half_t* w_eji     = (half_t*)alloc((size_t)128 * 128 * sizeof(half_t));
    half_t* w_ekj     = (half_t*)alloc((size_t)128 * 128 * sizeof(half_t));
    half_t* w_edown   = (half_t*)alloc((size_t)128 * 64 * sizeof(half_t));
    half_t* w_eup     = (half_t*)alloc((size_t)64 * 128 * sizeof(half_t));
    half_t* w_resb    = (half_t*)alloc((size_t)2 * 128 * 128 * sizeof(half_t));
    half_t* w_elin    = (half_t*)alloc((size_t)128 * 128 * sizeof(half_t));
    half_t* w_resa    = (half_t*)alloc((size_t)4 * 128 * 128 * sizeof(half_t));
    half_t* w_vup     = (half_t*)alloc((size_t)128 * 256 * sizeof(half_t));
    half_t* w_vlins   = (half_t*)alloc((size_t)3 * 256 * 256 * sizeof(half_t));
    // activations: edge tensors in f16, atomic targets in f32
    float*  dist    = (float*)alloc((size_t)E * sizeof(float));
    float*  rbf     = (float*)alloc((size_t)E * 6 * sizeof(float));
    half_t* rbf0h   = (half_t*)alloc((size_t)E * 128 * sizeof(half_t));  // + residual tmp
    half_t* e1      = (half_t*)alloc((size_t)E * 128 * sizeof(half_t));
    half_t* e1b     = (half_t*)alloc((size_t)E * 128 * sizeof(half_t));
    half_t* x_ji    = (half_t*)alloc((size_t)E * 128 * sizeof(half_t));
    half_t* x_kj    = (half_t*)alloc((size_t)E * 128 * sizeof(half_t));
    float*  e2buf   = (float*)alloc((size_t)E * 128 * sizeof(float));    // x_kj2 / e2
    float*  agg     = (float*)alloc((size_t)E * 64 * sizeof(float));
    float*  angle   = (float*)alloc((size_t)T * sizeof(float));
    unsigned int* torsu = (unsigned int*)alloc((size_t)T * sizeof(unsigned int));
    float*  nodeacc = (float*)alloc((size_t)N * 128 * sizeof(float));
    float*  v1      = (float*)alloc((size_t)N * 256 * sizeof(float));
    float*  v2      = (float*)alloc((size_t)N * 256 * sizeof(float));
    float*  vtot    = (float*)alloc((size_t)N * sizeof(float));
    (void)ws_size; (void)n_in; (void)dist;

    auto blocks1d = [](size_t n) -> int {
        size_t b = (n + 255) / 256;
        return b < 1 ? 1 : (int)b;
    };
    auto gemm_blocks = [](int rows) -> int {
        int mt = rows / 16;
        int b = (mt + 7) / 8;
        return b < 1 ? 1 : b;
    };
#define PACK(src, dst, Kk, Nn) \
    pack_w_kernel<<<blocks1d((size_t)(Kk) * (Nn)), 256, 0, stream>>>((src), (dst), (Kk), (Nn))
#define ZERO(p, n) \
    zero_kernel<<<blocks1d((size_t)(n)), 256, 0, stream>>>((p), (size_t)(n))
#define GEMM(Kk, SILU, BIAS, ADD, A, W, B_, AD, C_, R_, N_)                              \
    gemm_pk_kernel<Kk, SILU, BIAS, ADD><<<gemm_blocks(R_), 256, 0, stream>>>(            \
        (A), (W), (B_), (AD), (C_), (R_) / 16, (N_))

    // ---- 0) pack weights into WMMA fragment order (f16, ~0.9 MB total) ----
    PACK(p_initlin_W, w_initlin, 384, 128);
    PACK(p_e_ji_W, w_eji, 128, 128);
    PACK(p_e_kj_W, w_ekj, 128, 128);
    PACK(p_e_down, w_edown, 128, 64);
    PACK(p_e_up, w_eup, 64, 128);
    PACK(p_resb_W + 0 * 16384, w_resb + 0 * 16384, 128, 128);
    PACK(p_resb_W + 1 * 16384, w_resb + 1 * 16384, 128, 128);
    PACK(p_e_lin_W, w_elin, 128, 128);
    PACK(p_resa_W + 0 * 16384, w_resa + 0 * 16384, 128, 128);
    PACK(p_resa_W + 1 * 16384, w_resa + 1 * 16384, 128, 128);
    PACK(p_resa_W + 2 * 16384, w_resa + 2 * 16384, 128, 128);
    PACK(p_resa_W + 3 * 16384, w_resa + 3 * 16384, 128, 128);
    PACK(p_v_up_W, w_vup, 128, 256);
    PACK(p_v_lins_W + 0 * 65536, w_vlins + 0 * 65536, 256, 256);
    PACK(p_v_lins_W + 1 * 65536, w_vlins + 1 * 65536, 256, 256);
    PACK(p_v_lins_W + 2 * 65536, w_vlins + 2 * 65536, 256, 256);

    // ---- 1) geometry ----
    edge_geom_kernel<<<blocks1d(E), 256, 0, stream>>>(pos, edge_i, edge_j, dist, rbf, E);
    angle_kernel<<<blocks1d(T), 256, 0, stream>>>(pos, t_i, t_j, t_k, angle, torsu, T);
    torsion_kernel<<<blocks1d(Q), 256, 0, stream>>>(pos, q_i, q_j, q_k, q_kn, q_tri, torsu, Q);

    // ---- 2) init block ----
    rbf0h_kernel<<<blocks1d((size_t)E * 128), 256, 0, stream>>>(rbf, p_rbf0_W, p_rbf0_b, rbf0h, E);
    gemm_init_kernel<<<gemm_blocks(E), 256, 0, stream>>>(
        p_emb, x_atom, edge_i, edge_j, rbf0h, w_initlin, p_initlin_b, e1, E / 16);
    mulrbf_kernel<<<blocks1d((size_t)E * 128), 256, 0, stream>>>(rbf, p_initrbf1_W, e1, e2buf, E);

    // ---- 3) update_v #1 ----
    ZERO(vtot, N);
    ZERO(nodeacc, (size_t)N * 128);
    scatter_node_kernel<<<blocks1d((size_t)E * 128), 256, 0, stream>>>(e2buf, edge_i, nodeacc, E);
    GEMM(128, false, true, false, nodeacc, w_vup, p_v_up_b, (const float*)nullptr, v1, N, 256);
    GEMM(256, true, true, false, v1, w_vlins + 0 * 65536, p_v_lins_b + 0, (const float*)nullptr, v2, N, 256);
    GEMM(256, true, true, false, v2, w_vlins + 1 * 65536, p_v_lins_b + 256, (const float*)nullptr, v1, N, 256);
    GEMM(256, true, true, false, v1, w_vlins + 2 * 65536, p_v_lins_b + 512, (const float*)nullptr, v2, N, 256);
    vout_kernel<<<blocks1d(N), 256, 0, stream>>>(v2, p_v_out_W, vtot, N);

    // ---- 4) update_e ----
    GEMM(128, true, true, false, e1, w_eji, p_e_ji_b, (const half_t*)nullptr, x_ji, E, 128);
    GEMM(128, true, true, false, e1, w_ekj, p_e_kj_b, (const half_t*)nullptr, x_kj, E, 128);
    mulrbfh_kernel<<<blocks1d((size_t)E * 128), 256, 0, stream>>>(rbf, p_e_rbf1, p_e_rbf2, x_kj, E);
    GEMM(128, true, false, false, x_kj, w_edown, (const float*)nullptr, (const float*)nullptr,
         e2buf, E, 64);  // x_kj2 [E,64] f32
    ZERO(agg, (size_t)E * 64);
    triplet_kernel<<<blocks1d(T), 256, 0, stream>>>(
        angle, torsu, rbf, e2buf, idx_kj, idx_ji,
        p_e_sbf1, p_e_sbf2, p_e_t1, p_e_t2, agg, T);
    // e1b = silu(agg @ e_up) + x_ji
    GEMM(64, true, false, true, agg, w_eup, (const float*)nullptr, x_ji, e1b, E, 128);
    // residual block before skip
    GEMM(128, true, true, false, e1b, w_resb + 0 * 16384, p_resb_b + 0, (const half_t*)nullptr, rbf0h, E, 128);
    GEMM(128, true, true, true, rbf0h, w_resb + 1 * 16384, p_resb_b + 128, e1b, e1b, E, 128);
    // e1 = silu(e1b @ e_lin + b) + e1_in   (per-element read-then-write, safe)
    GEMM(128, true, true, true, e1b, w_elin, p_e_lin_b, e1, e1, E, 128);
    // two residual blocks after skip
    GEMM(128, true, true, false, e1, w_resa + 0 * 16384, p_resa_b + 0, (const half_t*)nullptr, rbf0h, E, 128);
    GEMM(128, true, true, true, rbf0h, w_resa + 1 * 16384, p_resa_b + 128, e1, e1, E, 128);
    GEMM(128, true, true, false, e1, w_resa + 2 * 16384, p_resa_b + 256, (const half_t*)nullptr, rbf0h, E, 128);
    GEMM(128, true, true, true, rbf0h, w_resa + 3 * 16384, p_resa_b + 384, e1, e1, E, 128);
    // e2 = (rbf @ e_rbf) * e1
    mulrbf_kernel<<<blocks1d((size_t)E * 128), 256, 0, stream>>>(rbf, p_e_rbf, e1, e2buf, E);

    // ---- 5) update_v #2 (accumulates into vtot) ----
    ZERO(nodeacc, (size_t)N * 128);
    scatter_node_kernel<<<blocks1d((size_t)E * 128), 256, 0, stream>>>(e2buf, edge_i, nodeacc, E);
    GEMM(128, false, true, false, nodeacc, w_vup, p_v_up_b, (const float*)nullptr, v1, N, 256);
    GEMM(256, true, true, false, v1, w_vlins + 0 * 65536, p_v_lins_b + 0, (const float*)nullptr, v2, N, 256);
    GEMM(256, true, true, false, v2, w_vlins + 1 * 65536, p_v_lins_b + 256, (const float*)nullptr, v1, N, 256);
    GEMM(256, true, true, false, v1, w_vlins + 2 * 65536, p_v_lins_b + 512, (const float*)nullptr, v2, N, 256);
    vout_kernel<<<blocks1d(N), 256, 0, stream>>>(v2, p_v_out_W, vtot, N);

    // ---- 6) readout ----
    ZERO(out, (size_t)out_size);
    batch_kernel<<<blocks1d(N), 256, 0, stream>>>(vtot, batch, out, N);

#undef PACK
#undef ZERO
#undef GEMM
}
